// DiscreteFlow_34780645163023
// MI455X (gfx1250) — compile-verified, matrix operand order checked by reading, not run on hardware
//
#include <hip/hip_runtime.h>
#include <hip/hip_bf16.h>

typedef __attribute__((ext_vector_type(16))) _Float16 v16h;
typedef __attribute__((ext_vector_type(8)))  float    v8f;

#define HDIM   1024
#define BATCH  32768
#define M_BLK  128
#define N_BLK  128
#define K_BLK  64
#define TS_STR 80   // halves per LDS row: 64 data + 16 pad -> 160B rows (32B-aligned, non-pow2)

__global__ __launch_bounds__(256) void init_out_kernel(float* __restrict__ out) {
  out[blockIdx.x * 256 + threadIdx.x] = 0.0f;
}

// Fold MADE masks into weights, transpose to N-major (WT[n][k]), cast to f16.
// Coalesced via 64x64 LDS transpose tiles. W0 padded from K=976 to 1024 with zeros.
__global__ __launch_bounds__(256) void prep_weights_kernel(
    const float* __restrict__ W0, const float* __restrict__ W1,
    const float* __restrict__ W2, _Float16* __restrict__ W0T,
    _Float16* __restrict__ W1T, _Float16* __restrict__ W2T) {
  __shared__ _Float16 tile[64][66];
  const int k0 = blockIdx.x * 64;
  const int n0 = blockIdx.y * 64;
  const int tid = threadIdx.x;
  const float* Ws[3] = {W0, W1, W2};
  _Float16* Ts[3] = {W0T, W1T, W2T};
  for (int w = 0; w < 3; ++w) {
    // load + mask: rows = k (contiguous n across lanes -> coalesced)
    for (int e = tid; e < 64 * 64; e += 256) {
      int r = e >> 6, c = e & 63;
      int k = k0 + r, n = n0 + c;
      float v = 0.0f;
      if (w == 0) {
        if (k < 976) {
          float m = 1.0f;
          if (k >= 16) { int l = (k - 16) >> 6; m = ((n & 15) >= (l + 1)) ? 1.0f : 0.0f; }
          v = Ws[0][k * HDIM + n] * m;
        }
      } else if (w == 1) {
        float m = ((n & 15) >= (k & 15)) ? 1.0f : 0.0f;
        v = Ws[1][k * HDIM + n] * m;
      } else {
        float m = ((n >> 6) >= (k & 15)) ? 1.0f : 0.0f;
        v = Ws[2][k * HDIM + n] * m;
      }
      tile[r][c] = (_Float16)v;
    }
    __syncthreads();
    // store transposed: rows = n (contiguous k across lanes -> coalesced)
    for (int e = tid; e < 64 * 64; e += 256) {
      int rr = e >> 6, cc = e & 63;
      Ts[w][(size_t)(n0 + rr) * HDIM + k0 + cc] = tile[cc][rr];
    }
    __syncthreads();
  }
}

// One GEMM shape for all three layers: (32768 x 1024) @ (1024 x 1024).
// A and B tiles double-buffered in LDS; B staged with global_load_async_to_lds
// and shared by all 8 waves (kills the 4x duplicated L2 stream).
// MODE 0: A = [c | one-hot(x)] built arithmetically in LDS; relu -> Hout (f16)
// MODE 1: A = Hin (async staged);                            relu -> Hout (f16)
// MODE 2: A = Hin (async staged); fused exp/group-sum/select -> atomicAdd(out)
template <int MODE>
__global__ __launch_bounds__(256) void made_gemm_kernel(
    const _Float16* __restrict__ Hin, const _Float16* __restrict__ WT,
    const float* __restrict__ bias, _Float16* __restrict__ Hout,
    float* __restrict__ out, const int* __restrict__ X,
    const float* __restrict__ Cin) {
  __shared__ __align__(32) _Float16 As[2][M_BLK][TS_STR];
  __shared__ __align__(32) _Float16 Bs[2][N_BLK][TS_STR];
  __shared__ int   xs[(MODE == 0) ? M_BLK : 1][16];
  __shared__ float cs[(MODE == 0) ? M_BLK : 1][16];

  const int tid   = threadIdx.x;
  const int lane  = tid & 31;
  const int wv    = tid >> 5;
  const int mw    = wv & 3;            // 4 M-waves
  const int nw    = wv >> 2;           // 2 N-waves
  const int m0    = blockIdx.x * M_BLK;
  const int n0    = blockIdx.y * N_BLK;
  const int mbase = m0 + mw * 32;      // wave tile: 32(M) x 64(N)
  const int nbase = n0 + nw * 64;
  const int lhalf = lane >> 4;         // 0/1
  const int l15   = lane & 15;

  if (MODE == 0) {
    for (int i = tid; i < M_BLK * 16; i += 256) {
      int r = i >> 4, cc = i & 15;
      xs[r][cc] = X[(m0 + r) * 16 + cc];
      cs[r][cc] = Cin[(m0 + r) * 16 + cc];
    }
    __syncthreads();
  }

  const int fr = tid >> 1;             // fill row 0..127
  const int fh = (tid & 1) * 32;       // fill k offset within row (halves)

  auto fillB = [&](int buf, int k0) {
    unsigned lds_addr = (unsigned)(unsigned long long)&Bs[buf][fr][fh];
    unsigned long long ga =
        (unsigned long long)&WT[(size_t)(n0 + fr) * HDIM + k0 + fh];
    asm volatile(
        "global_load_async_to_lds_b128 %0, %1, off\n\t"
        "global_load_async_to_lds_b128 %0, %1, off offset:16\n\t"
        "global_load_async_to_lds_b128 %0, %1, off offset:32\n\t"
        "global_load_async_to_lds_b128 %0, %1, off offset:48"
        :: "v"(lds_addr), "v"(ga) : "memory");
  };

  auto fillA = [&](int buf, int k0) {
    if (MODE == 0) {
      const int kg0 = k0 + fh;         // multiple of 32
      v16h z0, z1;
      #pragma unroll
      for (int i = 0; i < 16; ++i) { z0[i] = (_Float16)0.0f; z1[i] = (_Float16)0.0f; }
      if (kg0 == 0) {
        #pragma unroll
        for (int i = 0; i < 16; ++i) z0[i] = (_Float16)cs[fr][i];
      }
      *(v16h*)&As[buf][fr][fh]      = z0;
      *(v16h*)&As[buf][fr][fh + 16] = z1;
      // a 32-half window can straddle at most two one-hot groups
      int lo = (kg0 > 16) ? kg0 : 16;
      int hi = (kg0 + 32 < 976) ? (kg0 + 32) : 976;
      if (lo < hi) {
        int l0 = (lo - 16) >> 6;
        int l1 = (hi - 1 - 16) >> 6;
        for (int l = l0; l <= l1; ++l) {
          int pos = 16 + (l << 6) + xs[fr][l];       // global k of the single 1
          if (pos >= kg0 && pos < kg0 + 32)
            As[buf][fr][fh + (pos - kg0)] = (_Float16)1.0f;
        }
      }
    } else {
      unsigned lds_addr = (unsigned)(unsigned long long)&As[buf][fr][fh];
      unsigned long long ga =
          (unsigned long long)&Hin[(size_t)(m0 + fr) * HDIM + k0 + fh];
      asm volatile(
          "global_load_async_to_lds_b128 %0, %1, off\n\t"
          "global_load_async_to_lds_b128 %0, %1, off offset:16\n\t"
          "global_load_async_to_lds_b128 %0, %1, off offset:32\n\t"
          "global_load_async_to_lds_b128 %0, %1, off offset:48"
          :: "v"(lds_addr), "v"(ga) : "memory");
    }
  };
  auto fillWait = [&]() {
    asm volatile("s_wait_asynccnt 0x0" ::: "memory");
  };

  v8f acc[2][4];
  #pragma unroll
  for (int mt = 0; mt < 2; ++mt)
    #pragma unroll
    for (int nt = 0; nt < 4; ++nt)
      #pragma unroll
      for (int r = 0; r < 8; ++r) acc[mt][nt][r] = 0.0f;

  fillA(0, 0);
  fillB(0, 0);
  fillWait();
  __syncthreads();

  #pragma unroll 1
  for (int kb = 0; kb < HDIM / K_BLK; ++kb) {
    const int cur = kb & 1;
    if (kb + 1 < HDIM / K_BLK) {
      fillA(cur ^ 1, (kb + 1) * K_BLK);
      fillB(cur ^ 1, (kb + 1) * K_BLK);
    }

    #pragma unroll
    for (int ks = 0; ks < 2; ++ks) {
      const int ko = ks * 32;
      // Load ALL fragments for this k-step into distinct registers first so
      // the ds_loads batch into one clause and the 8 WMMAs issue back-to-back
      // behind a single wait (no per-pair s_wait_dscnt serialization).
      v16h a0 = *(const v16h*)&As[cur][mw * 32 + l15][ko + lhalf * 16];
      v16h a1 = *(const v16h*)&As[cur][mw * 32 + 16 + l15][ko + lhalf * 16];
      v16h bfr[4];
      #pragma unroll
      for (int nt = 0; nt < 4; ++nt)
        bfr[nt] = *(const v16h*)&Bs[cur][nw * 64 + nt * 16 + l15][ko + lhalf * 16];
      #pragma unroll
      for (int nt = 0; nt < 4; ++nt) {
        acc[0][nt] = __builtin_amdgcn_wmma_f32_16x16x32_f16(
            false, a0, false, bfr[nt], (short)0, acc[0][nt], false, false);
        acc[1][nt] = __builtin_amdgcn_wmma_f32_16x16x32_f16(
            false, a1, false, bfr[nt], (short)0, acc[1][nt], false, false);
      }
    }
    fillWait();
    __syncthreads();
  }

  if (MODE != 2) {
    // relu(acc + bias) -> f16 activation buffer
    #pragma unroll
    for (int nt = 0; nt < 4; ++nt) {
      int col = nbase + nt * 16 + l15;
      float bv = bias[col];
      #pragma unroll
      for (int mt = 0; mt < 2; ++mt) {
        int rowb = mbase + mt * 16 + lhalf * 8;
        #pragma unroll
        for (int r = 0; r < 8; ++r) {
          float h = acc[mt][nt][r] + bv;
          h = fmaxf(h, 0.0f);
          Hout[(size_t)(rowb + r) * HDIM + col] = (_Float16)h;
        }
      }
    }
  } else {
    // nw's 64 columns == exactly one softmax group l
    const int l = nbase >> 6;
    #pragma unroll
    for (int mt = 0; mt < 2; ++mt) {
      int rowb = mbase + mt * 16 + lhalf * 8;
      int xsel[8];
      #pragma unroll
      for (int r = 0; r < 8; ++r) xsel[r] = X[(rowb + r) * 16 + l];
      float s[8], ss[8];
      #pragma unroll
      for (int r = 0; r < 8; ++r) { s[r] = 0.0f; ss[r] = 0.0f; }
      #pragma unroll
      for (int nt = 0; nt < 4; ++nt) {
        int col = nbase + nt * 16 + l15;
        float bv = bias[col];
        int cl = nt * 16 + l15;      // column within the 64-wide group
        #pragma unroll
        for (int r = 0; r < 8; ++r) {
          float e = __expf(acc[mt][nt][r] + bv);
          s[r] += e;
          ss[r] += (cl == xsel[r]) ? e : 0.0f;
        }
      }
      #pragma unroll
      for (int r = 0; r < 8; ++r) {
        #pragma unroll
        for (int d = 1; d < 16; d <<= 1) {
          s[r]  += __shfl_xor(s[r],  d, 32);
          ss[r] += __shfl_xor(ss[r], d, 32);
        }
        if (l15 == 0)
          atomicAdd(&out[rowb + r], __logf(ss[r]) - __logf(s[r]));
      }
    }
  }
}

extern "C" void kernel_launch(void* const* d_in, const int* in_sizes, int n_in,
                              void* d_out, int out_size, void* d_ws, size_t ws_size,
                              hipStream_t stream) {
  const int*   X  = (const int*)d_in[0];
  const float* Cc = (const float*)d_in[1];
  const float* W0 = (const float*)d_in[2];
  const float* b0 = (const float*)d_in[3];
  const float* W1 = (const float*)d_in[4];
  const float* b1 = (const float*)d_in[5];
  const float* W2 = (const float*)d_in[6];
  const float* b2 = (const float*)d_in[7];
  float* out = (float*)d_out;

  char* ws = (char*)d_ws;
  _Float16* W0T = (_Float16*)(ws);
  _Float16* W1T = (_Float16*)(ws + (size_t)(2u << 20));
  _Float16* W2T = (_Float16*)(ws + (size_t)(4u << 20));
  _Float16* H1  = (_Float16*)(ws + (size_t)(6u << 20));            // 64 MiB
  _Float16* H2  = (_Float16*)(ws + (size_t)(70u << 20));           // 64 MiB

  init_out_kernel<<<BATCH / 256, 256, 0, stream>>>(out);
  {
    dim3 pgrid(HDIM / 64, HDIM / 64);
    prep_weights_kernel<<<pgrid, 256, 0, stream>>>(W0, W1, W2, W0T, W1T, W2T);
  }
  dim3 grid(BATCH / M_BLK, HDIM / N_BLK);
  made_gemm_kernel<0><<<grid, 256, 0, stream>>>(nullptr, W0T, b0, H1, nullptr, X, Cc);
  made_gemm_kernel<1><<<grid, 256, 0, stream>>>(H1, W1T, b1, H2, nullptr, nullptr, nullptr);
  made_gemm_kernel<2><<<grid, 256, 0, stream>>>(H2, W2T, b2, nullptr, out, X, nullptr);
}
